// mLSTM_10831907520820
// MI455X (gfx1250) — matrix operand order repaired and checked by reading, not compiled, not run on hardware
//
#include <hip/hip_runtime.h>

#define NB   8          // batch
#define SS   1024       // sequence
#define DD   256        // model dim
#define MM   (NB*SS)    // 8192 rows
#define CL   32         // chunk length for the chunkwise-parallel scan

typedef __bf16 bf16;
typedef __attribute__((ext_vector_type(8)))  bf16  v8bf;
typedef __attribute__((ext_vector_type(16))) bf16  v16bf;
typedef __attribute__((ext_vector_type(8)))  float v8f;

// ---------------- fragment helpers (ISA layouts, wave32) ----------------
// A-frag 16x32: lane = m + 16h; holds K chunks [8h,8h+8) and [16+8h,16+8h+8) of row m.
__device__ __forceinline__ v16bf ldfragA(const bf16* __restrict__ row, int h) {
    v8bf lo = *(const v8bf*)(row + 8 * h);
    v8bf hi = *(const v8bf*)(row + 16 + 8 * h);
    v16bf a;
    #pragma unroll
    for (int i = 0; i < 8; ++i) { a[i] = lo[i]; a[8 + i] = hi[i]; }
    return a;
}
// B-frag 32x16: lane = n + 16h; holds K chunks [8h,8h+8) and [16+8h,16+8h+8) of R[n][*] where B[k][n] = R[n][k].
__device__ __forceinline__ v16bf ldfragB(const bf16* __restrict__ row, int h) {
    v8bf lo = *(const v8bf*)(row + 8 * h);
    v8bf hi = *(const v8bf*)(row + 16 + 8 * h);
    v16bf b;
    #pragma unroll
    for (int i = 0; i < 8; ++i) { b[i] = lo[i]; b[8 + i] = hi[i]; }
    return b;
}
__device__ __forceinline__ v8f wmma_bf16(v16bf a, v16bf b, v8f c) {
    return __builtin_amdgcn_wmma_f32_16x16x32_bf16(false, a, false, b, (short)0, c, false, false);
}

// ---------------- f32 -> bf16 conversion ----------------
__global__ __launch_bounds__(256) void cvt_f32_bf16(const float* __restrict__ s,
                                                    bf16* __restrict__ d, int n) {
    int i = blockIdx.x * 256 + threadIdx.x;
    if (i < n) d[i] = (bf16)s[i];
}

// ---------------- WMMA GEMM: Y[m,n] = act( sum_k X[m,k]*W[n,k] + bias[n] ) ----------------
// Block = 256 threads = 8 waves laid out 4(M) x 2(N); block tile 64x128; grid (M/64, 2).
__global__ __launch_bounds__(256) void gemm_xwT(const bf16* __restrict__ X,
                                                const bf16* __restrict__ W,
                                                const float* __restrict__ bias,
                                                float* __restrict__ Y,
                                                int act) {
    const int K = DD, N = DD;
    const int lane = threadIdx.x & 31;
    const int wid  = threadIdx.x >> 5;
    const int wm = wid & 3, wn = wid >> 2;
    const int h = lane >> 4, l15 = lane & 15;
    const int row_a = blockIdx.x * 64 + wm * 16 + l15;
    const int col0  = blockIdx.y * 128 + wn * 64 + l15;

    v8f acc[4] = {};
    const bf16* xrow = X + (size_t)row_a * K;

    #pragma unroll
    for (int k0 = 0; k0 < K; k0 += 32) {
        v16bf a = ldfragA(xrow + k0, h);
        #pragma unroll
        for (int t = 0; t < 4; ++t) {
            v16bf bm = ldfragB(W + (size_t)(col0 + 16 * t) * K + k0, h);
            acc[t] = wmma_bf16(a, bm, acc[t]);
        }
    }
    const int row0 = blockIdx.x * 64 + wm * 16 + 8 * h;
    #pragma unroll
    for (int t = 0; t < 4; ++t) {
        int col = col0 + 16 * t;
        float bb = bias[col];
        #pragma unroll
        for (int j = 0; j < 8; ++j) {
            float v = acc[t][j] + bb;
            if (act == 1)      v = 1.0f / (1.0f + __expf(-v));   // sigmoid (forget gate)
            else if (act == 2) v = __expf(v);                    // exp (input gate)
            Y[(size_t)(row0 + j) * N + col] = v;
        }
    }
}

// ---------------- chunkwise-parallel mLSTM scan (WMMA) ----------------
// Per batch (1 workgroup, 256 thr = 8 waves), 32 serial chunks of L=32 steps.
// Within chunk:  Fcum[r][i] = prod f,  G[t][i] = i*v/Fcum,
//   num = Fcum .* ( tril(Q K^T) @ G  +  Q @ C_state^T )
//   C_state = diag(FcumL) (C_state + G^T @ K);   den via per-channel cumsum.
// LDS layout (bytes):
constexpr int OFF_FCUM = 256 * 256 * 2;            // Cs   bf16 [256][256] 128KB
constexpr int OFF_GT   = OFF_FCUM + CL * DD * 4;   // Fcum f32  [32][256]  32KB
constexpr int OFF_KB   = OFF_GT   + DD * CL * 2;   // Gt   bf16 [256][32]  16KB
constexpr int OFF_KT   = OFF_KB   + CL * DD * 2;   // Kb   bf16 [32][256]  16KB
constexpr int OFF_QB   = OFF_KT   + DD * CL * 2;   // Kt   bf16 [256][32]  16KB
constexpr int OFF_SA   = OFF_QB   + CL * DD * 2;   // Qb   bf16 [32][256]  16KB
constexpr int OFF_SAB  = OFF_SA   + CL * CL * 4;   // sA   f32  [32][32]    4KB
constexpr int OFF_DEN  = OFF_SAB  + CL * CL * 2;   // sAb  bf16 [32][32]    2KB
constexpr int OFF_WRED = OFF_DEN  + CL * 4;        // denv f32  [32]
constexpr int OFF_NST  = OFF_WRED + CL * 8 * 4;    // wred f32  [32][8]
constexpr int SMEM_CHUNK = OFF_NST + DD * 4;       // nst  f32  [256]

__global__ __launch_bounds__(256) void mlstm_chunk(const float* __restrict__ Fg,
                                                   const float* __restrict__ Ig,
                                                   const float* __restrict__ Kg,
                                                   const float* __restrict__ Vg,
                                                   const float* __restrict__ Qg,
                                                   bf16* __restrict__ Hbf) {
    extern __shared__ char smraw[];
    bf16*  Cs   = (bf16*)smraw;
    float* Fcum = (float*)(smraw + OFF_FCUM);
    bf16*  Gt   = (bf16*) (smraw + OFF_GT);
    bf16*  Kb   = (bf16*) (smraw + OFF_KB);
    bf16*  Kt   = (bf16*) (smraw + OFF_KT);
    bf16*  Qb   = (bf16*) (smraw + OFF_QB);
    float* sA   = (float*)(smraw + OFF_SA);
    bf16*  sAb  = (bf16*) (smraw + OFF_SAB);
    float* denv = (float*)(smraw + OFF_DEN);
    float* wred = (float*)(smraw + OFF_WRED);
    float* nst  = (float*)(smraw + OFF_NST);

    const int tid  = threadIdx.x;
    const int lane = tid & 31, w = tid >> 5;
    const int h = lane >> 4, l15 = lane & 15;
    const int b = blockIdx.x;

    for (int e = tid; e < DD * DD; e += 256) Cs[e] = (bf16)0.f;   // C_state = 0
    nst[tid] = 0.f;                                               // n_state = 0
    __syncthreads();

    const size_t base = (size_t)b * SS * DD;
    for (int c = 0; c < SS / CL; ++c) {
        const size_t cb = base + (size_t)c * CL * DD;

        // ---- Phase E: per-channel (i = k = tid) cumprod / transposes / den partials ----
        float Fc = 1.f, d = 0.f;
        const float ns0 = nst[tid];
        float pden[CL];
        #pragma unroll
        for (int r = 0; r < CL; ++r) {
            size_t idx = cb + (size_t)r * DD + tid;
            float f  = Fg[idx], ig = Ig[idx], kk = Kg[idx];
            float vv = Vg[idx], qv = Qg[idx];
            Fc *= f;
            Fcum[r * DD + tid] = Fc;
            Gt[tid * CL + r] = (bf16)(ig * vv / Fc);
            Kb[r * DD + tid] = (bf16)kk;
            Kt[tid * CL + r] = (bf16)kk;
            Qb[r * DD + tid] = (bf16)qv;
            d += ig * kk / Fc;
            pden[r] = qv * Fc * (ns0 + d);
        }
        nst[tid] = Fc * (ns0 + d);                 // only this thread touches nst[tid]
        // den reduction: sum pden[r] over 256 channels
        #pragma unroll
        for (int r = 0; r < CL; ++r) {
            float p = pden[r];
            #pragma unroll
            for (int off = 16; off > 0; off >>= 1) p += __shfl_down(p, off, 32);
            if (lane == 0) wred[r * 8 + w] = p;
        }
        __syncthreads();
        if (tid < CL) {
            float s = 0.f;
            #pragma unroll
            for (int j = 0; j < 8; ++j) s += wred[tid * 8 + j];
            s = fabsf(s);
            denv[tid] = s < 1.f ? 1.f : s;
        }
        __syncthreads();

        // ---- GEMM1: A = Q K^T (32x32), 4 waves, K swept 8x32 ----
        if (w < 4) {
            int mr = w & 1, nc = w >> 1;
            v8f acc = {};
            const bf16* arow = Qb + (size_t)(16 * mr + l15) * DD;
            const bf16* brow = Kb + (size_t)(16 * nc + l15) * DD;
            #pragma unroll
            for (int k0 = 0; k0 < DD; k0 += 32)
                acc = wmma_bf16(ldfragA(arow + k0, h), ldfragB(brow + k0, h), acc);
            #pragma unroll
            for (int j = 0; j < 8; ++j)
                sA[(16 * mr + j + 8 * h) * CL + 16 * nc + l15] = acc[j];
        }
        __syncthreads();
        // causal mask + bf16 cvt
        #pragma unroll
        for (int u = 0; u < 4; ++u) {
            int idx = tid * 4 + u, r = idx >> 5, tt = idx & 31;
            sAb[idx] = (bf16)(tt <= r ? sA[idx] : 0.f);
        }
        __syncthreads();

        // ---- GEMM2+3: num = tril(A)@G + Q@Cs^T (32x256); 32 tiles, 4/wave ----
        {
            int mr = w >> 2, wq = w & 3;
            v16bf aA = ldfragA(sAb + (size_t)(16 * mr + l15) * CL, h);
            v8f acc[4];
            #pragma unroll
            for (int u = 0; u < 4; ++u) {
                v8f z = {};
                int ncol = (wq * 4 + u) * 16;
                acc[u] = wmma_bf16(aA, ldfragB(Gt + (size_t)(ncol + l15) * CL, h), z);
            }
            const bf16* arow = Qb + (size_t)(16 * mr + l15) * DD;
            #pragma unroll
            for (int k0 = 0; k0 < DD; k0 += 32) {
                v16bf a = ldfragA(arow + k0, h);
                #pragma unroll
                for (int u = 0; u < 4; ++u) {
                    int ncol = (wq * 4 + u) * 16;
                    acc[u] = wmma_bf16(a, ldfragB(Cs + (size_t)(ncol + l15) * DD + k0, h), acc[u]);
                }
            }
            // h = Fcum .* num / den
            #pragma unroll
            for (int u = 0; u < 4; ++u) {
                int col = (wq * 4 + u) * 16 + l15;
                #pragma unroll
                for (int j = 0; j < 8; ++j) {
                    int r = 16 * mr + j + 8 * h;
                    float val = Fcum[r * DD + col] * acc[u][j] / denv[r];
                    Hbf[cb + (size_t)r * DD + col] = (bf16)val;
                }
            }
        }
        __syncthreads();   // all reads of Cs done before state update

        // ---- GEMM4: U = G^T @ K (256x256, K=32); Cs = FcumL .* (Cs + U); 32 tiles/wave ----
        #pragma unroll
        for (int mm = 0; mm < 2; ++mm) {
            int mi = 2 * w + mm;
            v16bf a = ldfragA(Gt + (size_t)(16 * mi + l15) * CL, h);
            for (int nc2 = 0; nc2 < 16; ++nc2) {
                v8f z = {};
                v8f acc = wmma_bf16(a, ldfragB(Kt + (size_t)(16 * nc2 + l15) * CL, h), z);
                int col = 16 * nc2 + l15;
                #pragma unroll
                for (int j = 0; j < 8; ++j) {
                    int row = 16 * mi + j + 8 * h;
                    float fl = Fcum[(CL - 1) * DD + row];
                    float cs = (float)Cs[(size_t)row * DD + col];
                    Cs[(size_t)row * DD + col] = (bf16)(fl * (cs + acc[j]));
                }
            }
        }
        __syncthreads();   // state + staging buffers stable before next chunk
    }
}

// ---------------- host launch ----------------
extern "C" void kernel_launch(void* const* d_in, const int* in_sizes, int n_in,
                              void* d_out, int out_size, void* d_ws, size_t ws_size,
                              hipStream_t stream) {
    (void)in_sizes; (void)n_in; (void)out_size; (void)ws_size;
    const float* x   = (const float*)d_in[0];
    const float* Wq  = (const float*)d_in[1];
    const float* bq  = (const float*)d_in[2];
    const float* Wk  = (const float*)d_in[3];
    const float* bk  = (const float*)d_in[4];
    // d_in[5]/d_in[6] (Wv, bv) unused — reference uses W_q for v
    const float* Wf  = (const float*)d_in[7];
    const float* bfv = (const float*)d_in[8];
    const float* Wi  = (const float*)d_in[9];
    const float* bi  = (const float*)d_in[10];
    const float* Wo  = (const float*)d_in[11];
    const float* bo  = (const float*)d_in[12];

    char* w = (char*)d_ws;
    bf16* xbf = (bf16*)w;  w += (size_t)MM * DD * sizeof(bf16);
    bf16* wqb = (bf16*)w;  w += (size_t)DD * DD * sizeof(bf16);
    bf16* wkb = (bf16*)w;  w += (size_t)DD * DD * sizeof(bf16);
    bf16* wfb = (bf16*)w;  w += (size_t)DD * DD * sizeof(bf16);
    bf16* wib = (bf16*)w;  w += (size_t)DD * DD * sizeof(bf16);
    bf16* wob = (bf16*)w;  w += (size_t)DD * DD * sizeof(bf16);
    float* qb = (float*)w; w += (size_t)MM * DD * sizeof(float);  // q (== v)
    float* kb = (float*)w; w += (size_t)MM * DD * sizeof(float);
    float* fb = (float*)w; w += (size_t)MM * DD * sizeof(float);
    float* ib = (float*)w; w += (size_t)MM * DD * sizeof(float);
    bf16* hbf = (bf16*)w;  w += (size_t)MM * DD * sizeof(bf16);

    cvt_f32_bf16<<<(MM * DD + 255) / 256, 256, 0, stream>>>(x,  xbf, MM * DD);
    cvt_f32_bf16<<<(DD * DD + 255) / 256, 256, 0, stream>>>(Wq, wqb, DD * DD);
    cvt_f32_bf16<<<(DD * DD + 255) / 256, 256, 0, stream>>>(Wk, wkb, DD * DD);
    cvt_f32_bf16<<<(DD * DD + 255) / 256, 256, 0, stream>>>(Wf, wfb, DD * DD);
    cvt_f32_bf16<<<(DD * DD + 255) / 256, 256, 0, stream>>>(Wi, wib, DD * DD);
    cvt_f32_bf16<<<(DD * DD + 255) / 256, 256, 0, stream>>>(Wo, wob, DD * DD);

    dim3 g(MM / 64, DD / 128);
    gemm_xwT<<<g, 256, 0, stream>>>(xbf, wqb, bq,  qb, 0);  // q (and v)
    gemm_xwT<<<g, 256, 0, stream>>>(xbf, wkb, bk,  kb, 0);  // k
    gemm_xwT<<<g, 256, 0, stream>>>(xbf, wfb, bfv, fb, 1);  // f = sigmoid
    gemm_xwT<<<g, 256, 0, stream>>>(xbf, wib, bi,  ib, 2);  // i = exp

    // chunkwise-parallel scan: 1 WGP per batch, C_state in LDS, WMMA inner chunks
    mlstm_chunk<<<NB, 256, SMEM_CHUNK, stream>>>(fb, ib, kb, qb, qb, hbf);

    // output projection
    gemm_xwT<<<g, 256, 0, stream>>>(hbf, wob, bo, (float*)d_out, 0);
}
